// MoEMLPFused_74191265071207
// MI455X (gfx1250) — matrix operand order repaired and checked by reading, not compiled, not run on hardware
//
#include <hip/hip_runtime.h>

// ---------------------------------------------------------------------------
// MoE MLP (swiglu) fused for gfx1250 / MI455X.
// Bucket token-expert pairs by expert, then 32-row expert tiles through two
// bf16 WMMA GEMMs (v_wmma_f32_16x16x32_bf16). 32-row tiles give 2x B-matrix
// reuse per weight fragment and hit the minimum 48MB weight traffic
// (L2-resident; ~2us at 23.3 TB/s if streamed from HBM). Swiglu pairing is
// done with lane shuffles (adjacent C-fragment columns), no LDS round trip.
// ---------------------------------------------------------------------------

typedef __attribute__((ext_vector_type(16))) __bf16 v16bf;
typedef __attribute__((ext_vector_type(8)))  float  v8f;

#define T_TOK  512
#define KSEL   2
#define NEXP   32
#define HDIM   512
#define IDIM   512
#define TWO_I  1024
#define NPAIR  (T_TOK * KSEL)

#define SWIGLU_ALPHA 1.702f
#define SWIGLU_LIMIT 7.0f

#define TILE_M    32          // block tile rows = 2 WMMA M-subtiles
#define MAX_TILES 32          // 32*32 = 1024 covers worst-case routing
#define H_STRIDE  520         // bf16 elems/row: 1040B -> 16B aligned, 4-bank rotate

union FragB { v16bf v; uint4 u[2]; };
union FragC { v8f  v; float f[8]; };

// A fragment (16x32 bf16, M x K). Lane L: m = L%16, hlf = L/16.
// VGPR0-3 hold K = hlf*8 + {0..7} (rel k0), VGPR4-7 hold K = 16 + hlf*8 + {0..7}.
__device__ __forceinline__ v16bf load_a(const __bf16* row, int k0, int hlf) {
    FragB f;
    f.u[0] = *reinterpret_cast<const uint4*>(row + k0 + hlf * 8);
    f.u[1] = *reinterpret_cast<const uint4*>(row + k0 + 16 + hlf * 8);
    return f.v;
}

// B fragment (32x16 bf16, K x N) from weights stored row-major (out_ch, k).
// Lane L: n = L%16 (caller picks column pointer), K = hlf*8 + {0..7} then
// K = 16 + hlf*8 + {0..7} (rel k0), matching load_a.
__device__ __forceinline__ v16bf load_b(const __bf16* wcol, int k0, int hlf) {
    FragB f;
    f.u[0] = *reinterpret_cast<const uint4*>(wcol + k0 + hlf * 8);
    f.u[1] = *reinterpret_cast<const uint4*>(wcol + k0 + 16 + hlf * 8);
    return f.v;
}

#define WMMA_BF16(A, B, C) \
    __builtin_amdgcn_wmma_f32_16x16x32_bf16(false, (A), false, (B), (short)0, (C), false, false)

// ------------------------------ support kernels ----------------------------

__global__ void moe_zero_kernel(float* __restrict__ acc, int n,
                                int* __restrict__ counts, int* __restrict__ cursor) {
    int i = blockIdx.x * blockDim.x + threadIdx.x;
    if (i < n) acc[i] = 0.0f;
    if (i < NEXP) { counts[i] = 0; cursor[i] = 0; }
}

__global__ void moe_count_kernel(const int* __restrict__ ei, int* __restrict__ counts) {
    int i = blockIdx.x * blockDim.x + threadIdx.x;
    if (i < NPAIR) atomicAdd(&counts[ei[i]], 1);
}

__global__ void moe_scan_kernel(const int* __restrict__ counts, int* __restrict__ offsets) {
    if (threadIdx.x == 0) {
        int s = 0;
        for (int e = 0; e < NEXP; ++e) { offsets[e] = s; s += counts[e]; }
        offsets[NEXP] = s;
    }
}

__global__ void moe_scatter_kernel(const int* __restrict__ ei, const int* __restrict__ offsets,
                                   int* __restrict__ cursor, int* __restrict__ bucket) {
    int i = blockIdx.x * blockDim.x + threadIdx.x;
    if (i < NPAIR) {
        int e = ei[i];
        int pos = offsets[e] + atomicAdd(&cursor[e], 1);
        bucket[pos] = i;                 // packed pair id: token = i>>1, slot k = i&1
    }
}

__global__ void moe_finalize_kernel(const float* __restrict__ acc, __bf16* __restrict__ out, int n) {
    int i = blockIdx.x * blockDim.x + threadIdx.x;
    if (i < n) out[i] = (__bf16)acc[i];
}

// ------------------------------- main kernel -------------------------------

__global__ __launch_bounds__(256)
void moe_expert_tile_kernel(const __bf16* __restrict__ x,
                            const __bf16* __restrict__ w1, const __bf16* __restrict__ b1,
                            const __bf16* __restrict__ w2, const __bf16* __restrict__ b2,
                            const __bf16* __restrict__ ew,
                            const int* __restrict__ counts, const int* __restrict__ offsets,
                            const int* __restrict__ bucket,
                            float* __restrict__ out_acc) {
    __shared__ __align__(16) __bf16 h_tile[TILE_M][H_STRIDE];
    __shared__ int   rows_s[TILE_M];
    __shared__ float roww_s[TILE_M];

    const int e    = blockIdx.y;
    const int tile = blockIdx.x;
    const int cnt  = counts[e];
    if (tile * TILE_M >= cnt) return;

    const int base = offsets[e];
    const int tid  = threadIdx.x;
    const int lane = tid & 31;
    const int wave = tid >> 5;
    const int hlf  = lane >> 4;      // lane half (0: lanes 0-15, 1: lanes 16-31)
    const int ln16 = lane & 15;

    // ---- stage row metadata (gathered tokens + routing weights) ----
    if (tid < TILE_M) {
        int r = tile * TILE_M + tid;
        if (r < cnt) {
            int packed  = bucket[base + r];
            rows_s[tid] = packed >> 1;
            roww_s[tid] = (float)ew[packed];
        } else {
            rows_s[tid] = -1;        // padded rows alias x row 0; outputs discarded
            roww_s[tid] = 0.0f;
        }
    }
    __syncthreads();

    // Per-lane gathered A-row pointers for the two M-subtiles (x is L2-resident).
    int tok0 = rows_s[ln16];
    int tok1 = rows_s[16 + ln16];
    const __bf16* xrow0 = x + (size_t)(tok0 < 0 ? 0 : tok0) * HDIM;
    const __bf16* xrow1 = x + (size_t)(tok1 < 0 ? 0 : tok1) * HDIM;

    const __bf16* w1e = w1 + (size_t)e * TWO_I * HDIM;
    const __bf16* b1e = b1 + e * TWO_I;

    // ---- GEMM1 (32 x 1024 <- 32x512 * 512x1024^T) + swiglu -> h_tile ----
    // Chunk c covers interleaved t-columns [32c, 32c+32): even->gate, odd->linear.
    for (int c = wave; c < IDIM / 16; c += 8) {
        FragC accA0, accA1, accB0, accB1;
#pragma unroll
        for (int i = 0; i < 8; ++i) {
            accA0.f[i] = 0.0f; accA1.f[i] = 0.0f;
            accB0.f[i] = 0.0f; accB1.f[i] = 0.0f;
        }

        const int colA = c * 32 + ln16;        // N-tile A: t-cols [32c, 32c+16)
        const int colB = c * 32 + 16 + ln16;   // N-tile B: t-cols [32c+16, 32c+32)
        const __bf16* wa = w1e + (size_t)colA * HDIM;
        const __bf16* wb = w1e + (size_t)colB * HDIM;

        for (int k0 = 0; k0 < HDIM; k0 += 32) {
            __builtin_prefetch(wa + k0 + 128, 0, 3);
            __builtin_prefetch(wb + k0 + 128, 0, 3);
            v16bf a0 = load_a(xrow0, k0, hlf);
            v16bf a1 = load_a(xrow1, k0, hlf);
            v16bf ba = load_b(wa, k0, hlf);
            v16bf bb = load_b(wb, k0, hlf);
            accA0.v = WMMA_BF16(a0, ba, accA0.v);
            accA1.v = WMMA_BF16(a1, ba, accA1.v);
            accB0.v = WMMA_BF16(a0, bb, accB0.v);
            accB1.v = WMMA_BF16(a1, bb, accB1.v);
        }

        // swiglu: C-fragment col = lane%16; gate/linear partner is the
        // adjacent lane -> one shuffle, even lanes store bf16 h to LDS.
        const float biasA = (float)b1e[colA];
        const float biasB = (float)b1e[colB];
        const int   jA = c * 16 + (ln16 >> 1);      // h col for tile A pairs
        const int   jB = jA + 8;                    // h col for tile B pairs
        FragC* accs[4] = { &accA0, &accA1, &accB0, &accB1 };
#pragma unroll
        for (int q = 0; q < 4; ++q) {
            const int   mt   = q & 1;               // M-subtile
            const bool  isB  = (q >= 2);
            const float bias = isB ? biasB : biasA;
            const int   hcol = isB ? jB : jA;
#pragma unroll
            for (int r = 0; r < 8; ++r) {
                float v = accs[q]->f[r] + bias;
                float p = __shfl_xor(v, 1, 32);     // partner channel
                if ((ln16 & 1) == 0) {
                    float tg = fminf(v, SWIGLU_LIMIT);
                    float tl = fminf(fmaxf(p, -SWIGLU_LIMIT), SWIGLU_LIMIT);
                    float sg = 1.0f / (1.0f + __expf(-SWIGLU_ALPHA * tg));
                    float h  = tg * sg * (tl + 1.0f);
                    h_tile[r + 8 * hlf + 16 * mt][hcol] = (__bf16)h;
                }
            }
        }
    }
    __syncthreads();

    const __bf16* w2e = w2 + (size_t)e * HDIM * IDIM;
    const __bf16* b2e = b2 + e * HDIM;

    // ---- GEMM2 (32 x 512 <- 32x512 * 512x512^T) + bias + weighted scatter ----
    for (int c = wave; c < HDIM / 16; c += 8) {
        FragC acc0, acc1;
#pragma unroll
        for (int i = 0; i < 8; ++i) { acc0.f[i] = 0.0f; acc1.f[i] = 0.0f; }

        const int col = c * 16 + ln16;
        const __bf16* wc = w2e + (size_t)col * IDIM;

        for (int k0 = 0; k0 < IDIM; k0 += 32) {
            __builtin_prefetch(wc + k0 + 128, 0, 3);
            v16bf a0 = load_a(&h_tile[ln16][0], k0, hlf);
            v16bf a1 = load_a(&h_tile[16 + ln16][0], k0, hlf);
            v16bf b  = load_b(wc, k0, hlf);
            acc0.v = WMMA_BF16(a0, b, acc0.v);
            acc1.v = WMMA_BF16(a1, b, acc1.v);
        }

        const float bias = (float)b2e[col];
#pragma unroll
        for (int mt = 0; mt < 2; ++mt) {
            FragC* acc = mt ? &acc1 : &acc0;
#pragma unroll
            for (int r = 0; r < 8; ++r) {
                int row = r + 8 * hlf + 16 * mt;
                int tok = rows_s[row];
                if (tok >= 0) {
                    atomicAdd(&out_acc[(size_t)tok * HDIM + col],
                              roww_s[row] * (acc->f[r] + bias));
                }
            }
        }
    }
}

// ------------------------------- launcher ----------------------------------

extern "C" void kernel_launch(void* const* d_in, const int* in_sizes, int n_in,
                              void* d_out, int out_size, void* d_ws, size_t ws_size,
                              hipStream_t stream) {
    (void)in_sizes; (void)n_in; (void)out_size; (void)ws_size;

    const __bf16* x  = (const __bf16*)d_in[0];
    const int*    ei = (const int*)d_in[1];
    const __bf16* ew = (const __bf16*)d_in[2];
    const __bf16* w1 = (const __bf16*)d_in[3];
    const __bf16* b1 = (const __bf16*)d_in[4];
    const __bf16* w2 = (const __bf16*)d_in[5];
    const __bf16* b2 = (const __bf16*)d_in[6];
    __bf16* out = (__bf16*)d_out;

    char* ws = (char*)d_ws;
    float* out_acc = (float*)ws;                            // 512*512 f32 = 1 MB
    int*   counts  = (int*)(ws + (size_t)T_TOK * HDIM * 4);
    int*   cursor  = counts + NEXP;
    int*   offsets = cursor + NEXP;                         // NEXP+1 entries
    int*   bucket  = offsets + NEXP + 1 + 2;                // NPAIR entries

    const int NOUT = T_TOK * HDIM;
    moe_zero_kernel<<<(NOUT + 255) / 256, 256, 0, stream>>>(out_acc, NOUT, counts, cursor);
    moe_count_kernel<<<(NPAIR + 255) / 256, 256, 0, stream>>>(ei, counts);
    moe_scan_kernel<<<1, 32, 0, stream>>>(counts, offsets);
    moe_scatter_kernel<<<(NPAIR + 255) / 256, 256, 0, stream>>>(ei, offsets, cursor, bucket);

    dim3 grid(MAX_TILES, NEXP);
    moe_expert_tile_kernel<<<grid, 256, 0, stream>>>(x, w1, b1, w2, b2, ew,
                                                     counts, offsets, bucket, out_acc);

    moe_finalize_kernel<<<(NOUT + 255) / 256, 256, 0, stream>>>(out_acc, out, NOUT);
}
